// MossAttention_9182640078968
// MI455X (gfx1250) — matrix operand order repaired and checked by reading, not compiled, not run on hardware
//
#include <hip/hip_runtime.h>
#include <cstdint>

// Problem constants (match reference)
#define WB   2
#define WS_  2048
#define WE   4096
#define WH   16
#define WD   256
#define ROT  64
#define WM_  (WB*WS_)      // 4096 rows of the token matrix
#define NQKV (3*WE)        // 12288

typedef __attribute__((ext_vector_type(16))) __bf16         v16bf;
typedef __attribute__((ext_vector_type(8)))  float          v8f;
typedef __attribute__((ext_vector_type(4)))  unsigned int   u32x4;
typedef __attribute__((ext_vector_type(4)))  unsigned short u16x4;
typedef __attribute__((ext_vector_type(4)))  float          f32x4;
typedef __attribute__((ext_vector_type(8)))  int            i32x8;
typedef __attribute__((ext_vector_type(4)))  int            i32x4;

union FragU { v16bf bf; u32x4 q[2]; };

__device__ inline unsigned short f32_to_bf16(float f) {
  union { float f; unsigned int u; } v; v.f = f;
  unsigned int u = v.u;
  u += 0x7FFFu + ((u >> 16) & 1u);   // round to nearest even
  return (unsigned short)(u >> 16);
}
__device__ inline float bf16_to_f32(unsigned short s) {
  union { unsigned int u; float f; } v; v.u = ((unsigned int)s) << 16; return v.f;
}

// A-fragment: 16x32 (MxK) bf16, row-major source with given element stride.
// Lanes 0-15 hold M=lane, K chunks {0..7,16..23}; lanes 16-31 hold {8..15,24..31}.
__device__ inline v16bf load_frag_a(const unsigned short* base, int stride, int lane) {
  int row = lane & 15;
  int k0  = (lane >> 4) << 3;
  FragU f;
  f.q[0] = *(const u32x4*)(base + row * stride + k0);
  f.q[1] = *(const u32x4*)(base + row * stride + k0 + 16);
  return f.bf;
}
// B-fragment: 32x16 (KxN) bf16 from N-major storage (row = n, contiguous k).
// Lanes 0-15 hold K chunks {0..7,16..23} of column n=lane; lanes 16-31 hold {8..15,24..31}.
__device__ inline v16bf load_frag_b(const unsigned short* base, int stride, int lane) {
  int col = lane & 15;
  int k0  = (lane >> 4) << 3;
  FragU f;
  f.q[0] = *(const u32x4*)(base + col * stride + k0);
  f.q[1] = *(const u32x4*)(base + col * stride + k0 + 16);
  return f.bf;
}

__device__ inline v8f wmma_bf16(v16bf a, v16bf b, v8f c) {
  return __builtin_amdgcn_wmma_f32_16x16x32_bf16(false, a, false, b, (short)0, c, false, false);
}

__device__ inline float hmax16(float v) {
  #pragma unroll
  for (int off = 1; off < 16; off <<= 1) v = fmaxf(v, __shfl_xor(v, off, 32));
  return v;
}
__device__ inline float hsum16(float v) {
  #pragma unroll
  for (int off = 1; off < 16; off <<= 1) v += __shfl_xor(v, off, 32);
  return v;
}

// ---------------------------------------------------------------------------
// Tensor Data Mover: 2-D tile load (bf16 elements) into LDS.
// D# packed per CDNA5 ISA 8.3/8.4: count=1, type=2("image"), data_size=1 (2B).
// tile_d0 elements per row (contiguous), tile_d1 rows, row stride = stride_d0.
// 6-arg builtin: (uint32x4 g0, int32x8 g1, int32x4 g2, int32x4 g3, int32x8, i32 cpol)
// ---------------------------------------------------------------------------
__device__ inline void tdm_load_2d(const unsigned short* gsrc, unsigned lds_byte_off,
                                   unsigned tensor_d0, unsigned tensor_d1,
                                   unsigned tile_d0, unsigned tile_d1,
                                   unsigned stride_d0) {
  unsigned long long ga = (unsigned long long)(uintptr_t)gsrc;
  u32x4 g0;
  g0[0] = 1u;                                           // count=1 (valid), user mode
  g0[1] = lds_byte_off;                                 // lds_addr (bytes)
  g0[2] = (unsigned)(ga & 0xFFFFFFFFu);                 // global_addr[31:0]
  g0[3] = (unsigned)((ga >> 32) & 0x01FFFFFFu) | (2u << 30);  // [56:32] | type=2
  i32x8 g1;
  g1[0] = (int)(1u << 16);                              // data_size=1 (2 bytes)
  g1[1] = (int)((tensor_d0 & 0xFFFFu) << 16);           // tensor_dim0[15:0]
  g1[2] = (int)(((tensor_d0 >> 16) & 0xFFFFu) | ((tensor_d1 & 0xFFFFu) << 16));
  g1[3] = (int)(((tensor_d1 >> 16) & 0xFFFFu) | ((tile_d0 & 0xFFFFu) << 16));
  g1[4] = (int)(tile_d1 & 0xFFFFu);                     // tile_dim1; tile_dim2=0
  g1[5] = (int)stride_d0;                               // tensor_dim0_stride[31:0]
  g1[6] = 0;
  g1[7] = 0;
  i32x4 z; z[0] = 0; z[1] = 0; z[2] = 0; z[3] = 0;
  i32x8 z8; z8[0] = 0; z8[1] = 0; z8[2] = 0; z8[3] = 0;
  z8[4] = 0; z8[5] = 0; z8[6] = 0; z8[7] = 0;
  __builtin_amdgcn_tensor_load_to_lds(g0, g1, z, z, z8, 0);
}
__device__ inline unsigned lds_off_of(const void* p) {
  return (unsigned)(uintptr_t)p;   // low 32 bits of flat address = LDS byte offset
}

// ---------------------------------------------------------------------------
// 1) fp32 -> bf16 elementwise (4 elements / thread, b128 loads)
// ---------------------------------------------------------------------------
__global__ __launch_bounds__(256) void conv_f32_bf16(const float* __restrict__ x,
                                                     unsigned short* __restrict__ y,
                                                     size_t n4) {
  size_t t = (size_t)blockIdx.x * blockDim.x + threadIdx.x;
  if (t >= n4) return;
  f32x4 v = *(const f32x4*)(x + t * 4);
  u16x4 o;
  o[0] = f32_to_bf16(v[0]); o[1] = f32_to_bf16(v[1]);
  o[2] = f32_to_bf16(v[2]); o[3] = f32_to_bf16(v[3]);
  *(u16x4*)(y + t * 4) = o;
}

// ---------------------------------------------------------------------------
// 2) fp32 [K,N] row-major -> bf16 transposed [N,K] (N-major weights for B-frags)
// ---------------------------------------------------------------------------
__global__ __launch_bounds__(256) void transpose_f32_bf16(const float* __restrict__ w,
                                                          unsigned short* __restrict__ wt,
                                                          int K, int N) {
  size_t t = (size_t)blockIdx.x * blockDim.x + threadIdx.x;
  if (t >= (size_t)K * N) return;
  size_t k = t % (size_t)K;
  size_t n = t / (size_t)K;
  wt[t] = f32_to_bf16(w[k * (size_t)N + n]);   // coalesced write along k
}

// ---------------------------------------------------------------------------
// 3/7) C[M,N] = A[M,K]_bf16 * BT[N,K]_bf16, fp32 accumulate via WMMA.
//      WG = 8 waves (4x2); WG tile 128x128, per-wave 32x64, K-step 64.
//      TDM double-buffered LDS staging: wave0 issues tensor_load_to_lds for
//      step i+2 while all waves compute step i; sync via TENSORcnt + barrier.
// ---------------------------------------------------------------------------
template <bool OUT_BF16>
__global__ __launch_bounds__(256) void gemm_bf16_wmma(const unsigned short* __restrict__ A,
                                                      const unsigned short* __restrict__ BT,
                                                      void* __restrict__ Cout,
                                                      int M, int N, int K) {
  __shared__ unsigned short lA[2][128 * 64];   // 16 KB per buffer
  __shared__ unsigned short lB[2][128 * 64];
  const int tid  = threadIdx.x;
  const int lane = tid & 31;
  const int wave = tid >> 5;
  const int wm   = wave >> 1;          // 0..3
  const int wn   = wave & 1;           // 0..1
  const int m0   = blockIdx.y * 128;
  const int n0   = blockIdx.x * 128;
  const int nsteps = K >> 6;           // K / 64

  v8f acc[2][4];
  #pragma unroll
  for (int i = 0; i < 2; ++i)
    #pragma unroll
    for (int j = 0; j < 4; ++j) acc[i][j] = (v8f)0.0f;

  // prologue: fill both buffers (async DMA, wave0 only; EXEC-independent issue)
  if (wave == 0) {
    tdm_load_2d(A  + (size_t)m0 * K, lds_off_of(&lA[0][0]), K, M, 64, 128, K);
    tdm_load_2d(BT + (size_t)n0 * K, lds_off_of(&lB[0][0]), K, N, 64, 128, K);
    if (nsteps > 1) {
      tdm_load_2d(A  + (size_t)m0 * K + 64, lds_off_of(&lA[1][0]), K, M, 64, 128, K);
      tdm_load_2d(BT + (size_t)n0 * K + 64, lds_off_of(&lB[1][0]), K, N, 64, 128, K);
    }
  }

  for (int i = 0; i < nsteps; ++i) {
    if (wave == 0) {
      if (i + 1 < nsteps) __builtin_amdgcn_s_wait_tensorcnt(2);  // pair i done
      else                __builtin_amdgcn_s_wait_tensorcnt(0);
    }
    __syncthreads();

    const unsigned short* pA = &lA[i & 1][0];
    const unsigned short* pB = &lB[i & 1][0];
    #pragma unroll
    for (int dc = 0; dc < 2; ++dc) {
      v16bf af[2], bfv[4];
      #pragma unroll
      for (int mi = 0; mi < 2; ++mi)
        af[mi] = load_frag_a(pA + (wm * 32 + mi * 16) * 64 + dc * 32, 64, lane);
      #pragma unroll
      for (int ni = 0; ni < 4; ++ni)
        bfv[ni] = load_frag_b(pB + (wn * 64 + ni * 16) * 64 + dc * 32, 64, lane);
      #pragma unroll
      for (int mi = 0; mi < 2; ++mi)
        #pragma unroll
        for (int ni = 0; ni < 4; ++ni)
          acc[mi][ni] = wmma_bf16(af[mi], bfv[ni], acc[mi][ni]);
    }
    __syncthreads();   // all waves done reading buffer i&1

    if (wave == 0 && i + 2 < nsteps) {
      int kn = (i + 2) << 6;
      tdm_load_2d(A  + (size_t)m0 * K + kn, lds_off_of(&lA[i & 1][0]), K, M, 64, 128, K);
      tdm_load_2d(BT + (size_t)n0 * K + kn, lds_off_of(&lB[i & 1][0]), K, N, 64, 128, K);
    }
  }

  const int mrb  = (lane >> 4) << 3;   // 0 or 8
  const int ncol = lane & 15;
  #pragma unroll
  for (int mi = 0; mi < 2; ++mi)
    #pragma unroll
    for (int ni = 0; ni < 4; ++ni)
      #pragma unroll
      for (int r = 0; r < 8; ++r) {
        size_t idx = (size_t)(m0 + wm * 32 + mi * 16 + mrb + r) * N
                   + (n0 + wn * 64 + ni * 16 + ncol);
        if (OUT_BF16) ((unsigned short*)Cout)[idx] = f32_to_bf16(acc[mi][ni][r]);
        else          ((float*)Cout)[idx]          = acc[mi][ni][r];
      }
}

// ---------------------------------------------------------------------------
// 4) rotary + MP head split + 1/sqrt(D) into Q; layouts:
//    Q,K: [B,H,S,D] row-major ; V: transposed [B,H,D,S] (for PV B-frags)
// ---------------------------------------------------------------------------
__global__ __launch_bounds__(256) void rotary_reorder(const unsigned short* __restrict__ qkv,
                                                      const int* __restrict__ pos_ids,
                                                      unsigned short* __restrict__ Qb,
                                                      unsigned short* __restrict__ Kb,
                                                      unsigned short* __restrict__ VTb) {
  size_t t = (size_t)blockIdx.x * blockDim.x + threadIdx.x; // B*S*H*(D/2) threads
  int d2 = (int)(t % (WD / 2));
  size_t r = t / (WD / 2);
  int h = (int)(r % WH);  r /= WH;
  int s = (int)(r % WS_);
  int b = (int)(r / WS_);
  int d = d2 * 2;
  int mp = h >> 2, sub = h & 3;                // H/MP = 4 heads per partition
  size_t rowbase = (size_t)(b * WS_ + s) * NQKV;
  int base = mp * (3 * WE / 4) + sub * WD + d; // 3*E/MP = 3072 per partition
  float q0 = bf16_to_f32(qkv[rowbase + base]);
  float q1 = bf16_to_f32(qkv[rowbase + base + 1]);
  float v0 = bf16_to_f32(qkv[rowbase + base + 1024]);
  float v1 = bf16_to_f32(qkv[rowbase + base + 1024 + 1]);
  float k0 = bf16_to_f32(qkv[rowbase + base + 2048]);
  float k1 = bf16_to_f32(qkv[rowbase + base + 2048 + 1]);

  if (d < ROT) {
    float pos  = (float)pos_ids[b * WS_ + s];
    float freq = __powf(10000.0f, -((float)d) / (float)ROT);
    float th   = pos * freq;
    float sn = __sinf(th), cs = __cosf(th);
    float nq0 = q0 * cs - q1 * sn, nq1 = q1 * cs + q0 * sn;
    float nk0 = k0 * cs - k1 * sn, nk1 = k1 * cs + k0 * sn;
    q0 = nq0; q1 = nq1; k0 = nk0; k1 = nk1;
  }
  const float inv_scale = 0.0625f;   // 1/sqrt(256)
  q0 *= inv_scale; q1 *= inv_scale;

  int bh = b * WH + h;
  size_t qk_idx = ((size_t)bh * WS_ + s) * WD + d;
  Qb[qk_idx]     = f32_to_bf16(q0);
  Qb[qk_idx + 1] = f32_to_bf16(q1);
  Kb[qk_idx]     = f32_to_bf16(k0);
  Kb[qk_idx + 1] = f32_to_bf16(k1);
  size_t vt0 = ((size_t)bh * WD + d) * WS_ + s;
  VTb[vt0]        = f32_to_bf16(v0);
  VTb[vt0 + WS_]  = f32_to_bf16(v1);
}

// ---------------------------------------------------------------------------
// 6) causal flash attention, workgroup-cooperative K/V staging via TDM.
//    One WG owns one (b,h) and 128 queries (8 waves x 16-row Q tiles).
//    Per 32-key step: K panel (32x256) and V panel (256x32) are DMA'd into
//    double-buffered LDS by wave0 (tensor_load_to_lds), two steps ahead of
//    the compute; waves read fragments with ds_load_b128.
//    Waves past their causal bound idle at the barriers (no wasted FLOPs).
// ---------------------------------------------------------------------------
__global__ __launch_bounds__(256) void flash_attn(const unsigned short* __restrict__ Qb,
                                                  const unsigned short* __restrict__ Kb,
                                                  const unsigned short* __restrict__ VTb,
                                                  unsigned short* __restrict__ attn_out) {
  __shared__ unsigned short ldsQ[8][16 * WD];   // 64 KB: per-wave Q tile
  __shared__ unsigned short ldsK[2][32 * WD];   // 32 KB: [key][d] panels
  __shared__ unsigned short ldsV[2][WD * 32];   // 32 KB: [d][s]  panels
  __shared__ unsigned short ldsP[8][16 * 32];   // 8 KB:  per-wave P tile
  const int tid  = threadIdx.x;
  const int wave = tid >> 5;
  const int lane = tid & 31;
  const int bh   = blockIdx.x >> 4;            // B*H = 32 ; 16 blocks per bh
  const int blk  = blockIdx.x & 15;
  const int Q0   = blk * 128;                  // 128 queries per block
  const int q0   = Q0 + wave * 16;             // this wave's 16-row tile
  const int kend = q0 + 16;                    // wave's causal bound (exclusive)
  const int nsteps = (Q0 + 128) >> 5;          // shared k-step count (32 keys each)

  const unsigned short* Qh = Qb  + (size_t)bh * WS_ * WD + (size_t)q0 * WD;
  const unsigned short* Kh = Kb  + (size_t)bh * WS_ * WD;
  const unsigned short* Vh = VTb + (size_t)bh * WD * WS_;

  // stage this wave's 16x256 Q tile into its LDS slice (wave-private)
  #pragma unroll
  for (int c = lane; c < 512; c += 32) {
    int row = c >> 5;
    int cc  = (c & 31) << 3;
    *(u32x4*)&ldsQ[wave][row * WD + cc] = *(const u32x4*)(Qh + (size_t)row * WD + cc);
  }

  v8f acc[16];
  #pragma unroll
  for (int nf = 0; nf < 16; ++nf) acc[nf] = (v8f)0.0f;
  float m[8], l[8];
  #pragma unroll
  for (int r = 0; r < 8; ++r) { m[r] = -1e30f; l[r] = 0.0f; }

  const int ncol = lane & 15;
  const int mrb  = (lane >> 4) << 3;

  // TDM prologue: K panel = rows kb..kb+31 of [S,D]; V panel = 256 rows x 32 s.
  if (wave == 0) {
    tdm_load_2d(Kh,      lds_off_of(&ldsK[0][0]), WD,  WS_, WD, 32, WD);
    tdm_load_2d(Vh,      lds_off_of(&ldsV[0][0]), WS_, WD,  32, WD, WS_);
    if (nsteps > 1) {
      tdm_load_2d(Kh + (size_t)32 * WD, lds_off_of(&ldsK[1][0]), WD,  WS_, WD, 32, WD);
      tdm_load_2d(Vh + 32,              lds_off_of(&ldsV[1][0]), WS_, WD,  32, WD, WS_);
    }
  }

  for (int i = 0; i < nsteps; ++i) {
    const int kb = i << 5;
    if (wave == 0) {
      if (i + 1 < nsteps) __builtin_amdgcn_s_wait_tensorcnt(2);  // pair i done
      else                __builtin_amdgcn_s_wait_tensorcnt(0);
    }
    __syncthreads();

    if (kb < kend) {   // wave-uniform causal guard; EXEC stays all-ones inside
      const unsigned short* pK = &ldsK[i & 1][0];
      const unsigned short* pV = &ldsV[i & 1][0];
      // ---- S = Q * K^T (two 16x16 fragments over 32 keys) ----
      v8f s0 = (v8f)0.0f, s1 = (v8f)0.0f;
      #pragma unroll
      for (int dc = 0; dc < 8; ++dc) {
        v16bf aq  = load_frag_a(&ldsQ[wave][dc * 32], WD, lane);
        v16bf kf0 = load_frag_b(pK + dc * 32, WD, lane);
        v16bf kf1 = load_frag_b(pK + 16 * WD + dc * 32, WD, lane);
        s0 = wmma_bf16(aq, kf0, s0);
        s1 = wmma_bf16(aq, kf1, s1);
      }
      // ---- causal mask ----
      #pragma unroll
      for (int r = 0; r < 8; ++r) {
        int q = q0 + mrb + r;
        if (kb + ncol > q)      s0[r] = -1e30f;
        if (kb + 16 + ncol > q) s1[r] = -1e30f;
      }
      // ---- online softmax; P (bf16) into per-wave LDS tile ----
      #pragma unroll
      for (int r = 0; r < 8; ++r) {
        float bm = hmax16(fmaxf(s0[r], s1[r]));
        float mn = fmaxf(m[r], bm);
        float corr = __expf(m[r] - mn);
        float p0 = __expf(s0[r] - mn);
        float p1 = __expf(s1[r] - mn);
        float rs = hsum16(p0 + p1);
        l[r] = l[r] * corr + rs;
        m[r] = mn;
        #pragma unroll
        for (int nf = 0; nf < 16; ++nf) acc[nf][r] *= corr;
        int Mr = mrb + r;
        ldsP[wave][Mr * 32 + ncol]      = f32_to_bf16(p0);
        ldsP[wave][Mr * 32 + ncol + 16] = f32_to_bf16(p1);
      }
      // ---- O += P * V (P 16x32 == one A-fragment) ----
      v16bf pf = load_frag_a(&ldsP[wave][0], 32, lane);
      #pragma unroll
      for (int nf = 0; nf < 16; ++nf) {
        v16bf vf = load_frag_b(pV + (nf * 16) * 32, 32, lane);
        acc[nf] = wmma_bf16(pf, vf, acc[nf]);
      }
    }
    __syncthreads();   // all waves done with buffer i&1

    if (wave == 0 && i + 2 < nsteps) {
      int kn = (i + 2) << 5;
      tdm_load_2d(Kh + (size_t)kn * WD, lds_off_of(&ldsK[i & 1][0]), WD,  WS_, WD, 32, WD);
      tdm_load_2d(Vh + kn,              lds_off_of(&ldsV[i & 1][0]), WS_, WD,  32, WD, WS_);
    }
  }

  // normalize and store bf16 [B*S, E] row-major (col = h*D + d)
  const int b = bh >> 4;
  const int h = bh & 15;
  #pragma unroll
  for (int r = 0; r < 8; ++r) {
    float inv = 1.0f / l[r];
    int row = b * WS_ + q0 + mrb + r;
    #pragma unroll
    for (int nf = 0; nf < 16; ++nf) {
      size_t idx = (size_t)row * WE + h * WD + nf * 16 + ncol;
      attn_out[idx] = f32_to_bf16(acc[nf][r] * inv);
    }
  }
}

// ---------------------------------------------------------------------------
// host-side orchestration
// ---------------------------------------------------------------------------
extern "C" void kernel_launch(void* const* d_in, const int* in_sizes, int n_in,
                              void* d_out, int out_size, void* d_ws, size_t ws_size,
                              hipStream_t stream) {
  (void)in_sizes; (void)n_in; (void)out_size; (void)ws_size;
  const float* hs   = (const float*)d_in[0];   // [B,S,E]
  const float* wqkv = (const float*)d_in[1];   // [E, 3E]
  const float* wout = (const float*)d_in[2];   // [E, E]
  const int*   pos  = (const int*)d_in[3];     // [B,S]
  float* out = (float*)d_out;                  // [B,S,E] fp32

  char* ws = (char*)d_ws;
  size_t o = 0;
  auto take = [&](size_t bytes) { char* p = ws + o; o += (bytes + 255) & ~(size_t)255; return p; };

  unsigned short* Xb     = (unsigned short*)take((size_t)WM_ * WE * 2);     // 33.5 MB
  unsigned short* WqkvT  = (unsigned short*)take((size_t)NQKV * WE * 2);    // 100.7 MB
  unsigned short* QKVb   = (unsigned short*)take((size_t)WM_ * NQKV * 2);   // 100.7 MB
  unsigned short* Qb     = (unsigned short*)take((size_t)WM_ * WE * 2);
  unsigned short* Kb     = (unsigned short*)take((size_t)WM_ * WE * 2);
  unsigned short* VTb    = (unsigned short*)take((size_t)WM_ * WE * 2);
  // dead regions reused once their consumers have run (in-stream order):
  unsigned short* WoT    = WqkvT;  // w_out^T overwrites w_qkv^T after GEMM1
  unsigned short* attnb  = Xb;     // attention output overwrites X after GEMM1

  // 1) X -> bf16
  {
    size_t n4 = (size_t)WM_ * WE / 4;
    conv_f32_bf16<<<(unsigned)(n4 / 256), 256, 0, stream>>>(hs, Xb, n4);
  }
  // 2) w_qkv^T -> bf16 (N-major)
  {
    size_t n = (size_t)WE * NQKV;
    transpose_f32_bf16<<<(unsigned)((n + 255) / 256), 256, 0, stream>>>(wqkv, WqkvT, WE, NQKV);
  }
  // 3) QKV GEMM (bf16 out)
  gemm_bf16_wmma<true><<<dim3(NQKV / 128, WM_ / 128), 256, 0, stream>>>(
      Xb, WqkvT, QKVb, WM_, NQKV, WE);
  // 4) rotary + reorder
  {
    size_t n = (size_t)WB * WS_ * WH * (WD / 2);
    rotary_reorder<<<(unsigned)(n / 256), 256, 0, stream>>>(QKVb, pos, Qb, Kb, VTb);
  }
  // 5) w_out^T -> bf16 (reuses WqkvT region)
  {
    size_t n = (size_t)WE * WE;
    transpose_f32_bf16<<<(unsigned)((n + 255) / 256), 256, 0, stream>>>(wout, WoT, WE, WE);
  }
  // 6) flash attention: one block per (b,h, 128-query slab)
  flash_attn<<<WB * WH * (WS_ / 128), 256, 0, stream>>>(Qb, Kb, VTb, attnb);
  // 7) output projection (fp32 out)
  gemm_bf16_wmma<false><<<dim3(WE / 128, WM_ / 128), 256, 0, stream>>>(
      attnb, WoT, out, WM_, WE, WE);
}
